// AFGRL_58995670778172
// MI455X (gfx1250) — compile-verified, matrix operand order checked by reading, not run on hardware
//
#include <hip/hip_runtime.h>
#include <hip/hip_bf16.h>

// ---------------- problem constants ----------------
#define ND        16384
#define DD        256
#define TK        8
#define NRUNS     5
#define NC        100
#define NCP       112          // centroids padded to 7 * 16
#define NITERS    20
#define NE        262144
#define HASH_CAP  (1u << 19)   // 2x the edge count, open addressing
#define HASH_MASK (HASH_CAP - 1u)
#define EMPTY_SLOT 0xFFFFFFFFu

typedef __attribute__((ext_vector_type(16))) _Float16 v16h;
typedef __attribute__((ext_vector_type(8)))  _Float16 v8h;
typedef __attribute__((ext_vector_type(8)))  float    v8f;

// Build a 16-half WMMA operand fragment for the 16x16x32 f16 op.
// Per CDNA5 ISA (16-bit A-matrix 16x32): lanes 0-15 hold K = {kb..kb+7, kb+16..kb+23}
// with kb=0; lanes 16-31 same with kb=8.  Two contiguous 8-half (16B) runs.
__device__ inline v16h load_frag(const _Float16* rowp, int k0, int kb) {
  v8h lo = *(const v8h*)(rowp + k0 + kb);
  v8h hi = *(const v8h*)(rowp + k0 + kb + 16);
  return __builtin_shufflevector(lo, hi, 0,1,2,3,4,5,6,7,8,9,10,11,12,13,14,15);
}

// ---------------- 1) L2 normalize, emit f16 operands ----------------
__global__ void norm_kernel(const float* __restrict__ stu,
                            const float* __restrict__ tea,
                            _Float16* __restrict__ sf,
                            _Float16* __restrict__ tf,
                            float* __restrict__ inv_t) {
  int row = blockIdx.x;
  int which = blockIdx.y;            // 0 = student, 1 = teacher
  const float* src = which ? tea : stu;
  const float* p = src + (size_t)row * DD;
  float v[4]; float ss = 0.f;
  #pragma unroll
  for (int e = 0; e < 4; ++e) { v[e] = p[threadIdx.x * 4 + e]; ss += v[e] * v[e]; }
  __shared__ float red[64];
  red[threadIdx.x] = ss; __syncthreads();
  for (int off = 32; off > 0; off >>= 1) {
    if ((int)threadIdx.x < off) red[threadIdx.x] += red[threadIdx.x + off];
    __syncthreads();
  }
  float inv = 1.0f / fmaxf(sqrtf(red[0]), 1e-12f);
  _Float16* dst = which ? tf : sf;
  #pragma unroll
  for (int e = 0; e < 4; ++e)
    dst[(size_t)row * DD + threadIdx.x * 4 + e] = (_Float16)(v[e] * inv);
  if (which && threadIdx.x == 0) inv_t[row] = inv;
}

// ---------------- 2) fused WMMA GEMM (s @ t^T) + diag(+10) + top-8 ----------------
// One workgroup per 16 rows; 8 waves stripe the 1024 column tiles.
// B operand is double-buffered in registers (ping-pong, unroll-by-2) so a full
// clause of 16 b128 loads overlaps the previous tile's 8 WMMAs + top-k scan.
__global__ __launch_bounds__(256) void gemm_topk_kernel(
    const _Float16* __restrict__ sf, const _Float16* __restrict__ tf,
    int* __restrict__ iknn, float* __restrict__ out) {
  __shared__ float tile[8][16][16];     // one 16x16 f32 tile per wave
  __shared__ float mv[16][16][8];       // [row][list][k] merge values
  __shared__ int   mi[16][16][8];

  int tid  = threadIdx.x;
  int wave = tid >> 5;
  int lane = tid & 31;
  int lrow = lane & 15;
  int chalf = lane >> 4;                // column half for the scan phase
  int kb   = chalf ? 8 : 0;
  int i0   = blockIdx.x * 16;

  // A fragments for this row block: held in registers for the whole sweep.
  v16h af[8];
  {
    const _Float16* arow = sf + (size_t)(i0 + lrow) * DD;
    #pragma unroll
    for (int kk = 0; kk < 8; ++kk) af[kk] = load_frag(arow, kk * 32, kb);
  }

  float tv[TK]; int ti[TK];
  #pragma unroll
  for (int q = 0; q < TK; ++q) { tv[q] = -1e30f; ti[q] = 0; }

  auto load_btile = [&](int j0, v16h* bf) {
    const _Float16* brow = tf + (size_t)(j0 + lrow) * DD;
    #pragma unroll
    for (int kk = 0; kk < 8; ++kk) bf[kk] = load_frag(brow, kk * 32, kb);
  };

  auto process_tile = [&](int j0, const v16h* bf) {
    v8f acc = {};
    #pragma unroll
    for (int kk = 0; kk < 8; ++kk)
      acc = __builtin_amdgcn_wmma_f32_16x16x32_f16(false, af[kk], false, bf[kk],
                                                   (short)0, acc, false, false);
    int mbase = chalf ? 8 : 0;
    #pragma unroll
    for (int r = 0; r < 8; ++r) tile[wave][mbase + r][lrow] = acc[r];
    __syncthreads();
    int i = i0 + lrow;
    #pragma unroll
    for (int cc = 0; cc < 8; ++cc) {
      int c = chalf * 8 + cc;
      int j = j0 + c;
      float v = tile[wave][lrow][c];
      if (i == j) v += 10.0f;
      if (v > tv[TK - 1]) {
        float cv = v; int ci = j;
        #pragma unroll
        for (int p = 0; p < TK; ++p) {
          bool g = cv > tv[p];
          float nv = g ? cv : tv[p]; int ni = g ? ci : ti[p];
          cv = g ? tv[p] : cv;       ci = g ? ti[p] : ci;
          tv[p] = nv; ti[p] = ni;
        }
      }
    }
    __syncthreads();
  };

  // Ping-pong over column tiles: each wave owns tiles {wave + 8*t}, 128 total,
  // processed two per outer iteration (64 iterations, identical barrier counts
  // across all waves).
  v16h b0[8], b1[8];
  load_btile(wave * 16, b0);
  #define NOUTER (ND / 16 / 16)         // 64
  for (int it = 0; it < NOUTER; ++it) {
    int jt0 = wave + it * 16;
    int jt1 = jt0 + 8;
    // prefetch two tiles ahead (gfx1250 global_prefetch_b8), then fill b1
    __builtin_prefetch(tf + (size_t)(((jt1 + 8) & (ND / 16 - 1)) * 16 + lrow) * DD, 0, 1);
    load_btile(jt1 * 16, b1);
    process_tile(jt0 * 16, b0);
    if (it + 1 < NOUTER) load_btile((jt0 + 16) * 16, b0);
    process_tile(jt1 * 16, b1);
  }

  // dump per-lane lists: 16 lists per row (8 waves x 2 halves)
  int list = wave * 2 + chalf;
  #pragma unroll
  for (int q = 0; q < TK; ++q) { mv[lrow][list][q] = tv[q]; mi[lrow][list][q] = ti[q]; }
  __syncthreads();

  if (tid < 16) {
    float fv[TK]; int fi[TK];
    #pragma unroll
    for (int q = 0; q < TK; ++q) { fv[q] = -1e30f; fi[q] = 0; }
    for (int l = 0; l < 16; ++l)
      for (int q = 0; q < TK; ++q) {
        float v = mv[tid][l][q]; int j = mi[tid][l][q];
        if (v > fv[TK - 1]) {
          float cv = v; int ci = j;
          #pragma unroll
          for (int p = 0; p < TK; ++p) {
            bool g = cv > fv[p];
            float nv = g ? cv : fv[p]; int ni = g ? ci : fi[p];
            cv = g ? fv[p] : cv;       ci = g ? fi[p] : ci;
            fv[p] = nv; fi[p] = ni;
          }
        }
      }
    int i = i0 + tid;
    #pragma unroll
    for (int q = 0; q < TK; ++q) {
      iknn[i * TK + q] = fi[q];
      out[i * TK + q] = (float)fi[q];                    // region 0: I_knn
      out[2 * (ND * TK) + i * TK + q] = fv[q];           // region 2: topk_vals
    }
  }
}

// ---------------- 3) k-means ----------------
__global__ void km_init_kernel(const _Float16* __restrict__ tf,
                               float* __restrict__ cent, int run) {
  int c = blockIdx.x, d = threadIdx.x;       // 112 blocks x 256 threads
  float v = 0.f;
  if (c < NC) {
    unsigned idx = ((unsigned)c * 977u + (unsigned)run * 4999u + 131u) % ND;
    v = (float)tf[(size_t)idx * DD + d];
  }
  cent[c * DD + d] = v;
}

__global__ void km_prep_kernel(const float* __restrict__ cent,
                               _Float16* __restrict__ centh,
                               float* __restrict__ csq) {
  int c = blockIdx.x, t = threadIdx.x;       // 112 blocks x 64 threads
  float ss = 0.f;
  #pragma unroll
  for (int e = 0; e < 4; ++e) {
    float v = cent[c * DD + t * 4 + e];
    ss += v * v;
    centh[c * DD + t * 4 + e] = (_Float16)v;
  }
  __shared__ float red[64];
  red[t] = ss; __syncthreads();
  for (int off = 32; off > 0; off >>= 1) {
    if (t < off) red[t] += red[t + off];
    __syncthreads();
  }
  if (t == 0) csq[c] = (c < NC) ? red[0] : 1e30f;   // kill padded centroids
}

// argmin dist  ==  argmax (x . c - 0.5*|c|^2); WMMA over 7 column tiles
__global__ __launch_bounds__(256) void km_assign_kernel(
    const _Float16* __restrict__ tf, const _Float16* __restrict__ centh,
    const float* __restrict__ csq, int* __restrict__ assign_out) {
  __shared__ float sc[16][NCP];
  int tid = threadIdx.x, wave = tid >> 5, lane = tid & 31;
  int lrow = lane & 15, kb = (lane < 16) ? 0 : 8;
  int i0 = blockIdx.x * 16;
  if (wave < 7) {
    const _Float16* arow = tf + (size_t)(i0 + lrow) * DD;
    const _Float16* brow = centh + (size_t)(wave * 16 + lrow) * DD;
    // batch all operand loads into one clause, then back-to-back WMMAs
    v16h af[8], bf[8];
    #pragma unroll
    for (int kk = 0; kk < 8; ++kk) { af[kk] = load_frag(arow, kk * 32, kb);
                                     bf[kk] = load_frag(brow, kk * 32, kb); }
    v8f acc = {};
    #pragma unroll
    for (int kk = 0; kk < 8; ++kk)
      acc = __builtin_amdgcn_wmma_f32_16x16x32_f16(false, af[kk], false, bf[kk],
                                                   (short)0, acc, false, false);
    int mbase = (lane < 16) ? 0 : 8;
    #pragma unroll
    for (int r = 0; r < 8; ++r) sc[mbase + r][wave * 16 + lrow] = acc[r];
  }
  __syncthreads();
  if (tid < 16) {
    int best = 0; float bv = -1e30f;
    for (int c = 0; c < NCP; ++c) {
      float v = sc[tid][c] - 0.5f * csq[c];
      if (v > bv) { bv = v; best = c; }
    }
    assign_out[i0 + tid] = best;
  }
}

__global__ void km_clear_kernel(float* __restrict__ sums, float* __restrict__ cnts) {
  int g = blockIdx.x * 256 + threadIdx.x;
  if (g < NC * DD) sums[g] = 0.f;
  if (g < 128) cnts[g] = 0.f;
}

__global__ void km_accum_kernel(const float* __restrict__ tea,
                                const float* __restrict__ inv_t,
                                const int* __restrict__ assign,
                                float* __restrict__ sums, float* __restrict__ cnts) {
  int g = blockIdx.x * 256 + threadIdx.x;   // ND*4 threads
  int i = g >> 2, ch = g & 3;
  int a = assign[i];
  float inv = inv_t[i];
  const float* x = tea + (size_t)i * DD + ch * 64;
  float* s = sums + a * DD + ch * 64;
  for (int d = 0; d < 64; ++d) atomicAdd(&s[d], x[d] * inv);
  if (ch == 0) atomicAdd(&cnts[a], 1.0f);
}

__global__ void km_update_kernel(float* __restrict__ cent,
                                 const float* __restrict__ sums,
                                 const float* __restrict__ cnts) {
  int g = blockIdx.x * 256 + threadIdx.x;
  if (g >= NC * DD) return;
  float cnt = cnts[g >> 8];
  if (cnt > 0.f) cent[g] = sums[g] / fmaxf(cnt, 1.f);
}

// ---------------- 4) edge hash set ----------------
__global__ void hash_clear_kernel(unsigned* __restrict__ ht) {
  unsigned g = blockIdx.x * 256 + threadIdx.x;
  if (g < HASH_CAP) ht[g] = EMPTY_SLOT;
}

__global__ void hash_insert_kernel(const int* __restrict__ ei, unsigned* __restrict__ ht) {
  int g = blockIdx.x * 256 + threadIdx.x;
  if (g >= NE) return;
  unsigned s = (unsigned)ei[g], d = (unsigned)ei[NE + g];
  unsigned code = (s << 14) | d;          // n=16384 -> 28-bit codes
  unsigned h = (code * 2654435761u) & HASH_MASK;
  for (;;) {
    unsigned prev = atomicCAS(&ht[h], EMPTY_SLOT, code);
    if (prev == EMPTY_SLOT || prev == code) break;
    h = (h + 1) & HASH_MASK;
  }
}

// ---------------- 5) pos_mask = in_adj | agree ----------------
__global__ void finalize_kernel(const int* __restrict__ iknn,
                                const int* __restrict__ labels,
                                const unsigned* __restrict__ ht,
                                float* __restrict__ out) {
  int g = blockIdx.x * 256 + threadIdx.x;
  if (g >= ND * TK) return;
  int i = g >> 3;
  int j = iknn[g];
  bool agree = false;
  #pragma unroll
  for (int r = 0; r < NRUNS; ++r)
    agree |= (labels[r * ND + i] == labels[r * ND + j]);
  unsigned code = ((unsigned)i << 14) | (unsigned)j;
  unsigned h = (code * 2654435761u) & HASH_MASK;
  bool found = false;
  for (;;) {
    unsigned v = ht[h];
    if (v == code) { found = true; break; }
    if (v == EMPTY_SLOT) break;
    h = (h + 1) & HASH_MASK;
  }
  out[ND * TK + g] = (found || agree) ? 1.0f : 0.0f;     // region 1: pos_mask
}

// ---------------- launcher ----------------
extern "C" void kernel_launch(void* const* d_in, const int* in_sizes, int n_in,
                              void* d_out, int out_size, void* d_ws, size_t ws_size,
                              hipStream_t stream) {
  (void)in_sizes; (void)n_in; (void)out_size; (void)ws_size;
  const float* stu = (const float*)d_in[0];
  const float* tea = (const float*)d_in[1];
  const int*   ei  = (const int*)d_in[2];
  float* out = (float*)d_out;

  // workspace layout (bytes)
  char* ws = (char*)d_ws;
  size_t off = 0;
  _Float16* tf    = (_Float16*)(ws + off); off += (size_t)ND * DD * 2;     // 8 MB
  _Float16* sf    = (_Float16*)(ws + off); off += (size_t)ND * DD * 2;     // 8 MB
  float*    inv_t = (float*)(ws + off);    off += (size_t)ND * 4;
  int*      iknn  = (int*)(ws + off);      off += (size_t)ND * TK * 4;
  int*      labels= (int*)(ws + off);      off += (size_t)NRUNS * ND * 4;
  float*    cent  = (float*)(ws + off);    off += (size_t)NCP * DD * 4;
  _Float16* centh = (_Float16*)(ws + off); off += (size_t)NCP * DD * 2;
  float*    csq   = (float*)(ws + off);    off += 512;
  float*    sums  = (float*)(ws + off);    off += (size_t)NC * DD * 4;
  float*    cnts  = (float*)(ws + off);    off += 512;
  int*      assign= (int*)(ws + off);      off += (size_t)ND * 4;
  unsigned* ht    = (unsigned*)(ws + off); off += (size_t)HASH_CAP * 4;    // 2 MB

  // 1) normalize both inputs -> f16 WMMA operands (+ inv norms for teacher)
  norm_kernel<<<dim3(ND, 2), 64, 0, stream>>>(stu, tea, sf, tf, inv_t);

  // 2) fused GEMM + top-k (writes out regions 0 and 2)
  gemm_topk_kernel<<<ND / 16, 256, 0, stream>>>(sf, tf, iknn, out);

  // 4) edge hash set (independent of k-means)
  hash_clear_kernel<<<HASH_CAP / 256, 256, 0, stream>>>(ht);
  hash_insert_kernel<<<NE / 256, 256, 0, stream>>>(ei, ht);

  // 3) k-means x NRUNS (deterministic init; jax PRNG choice not reproducible)
  for (int run = 0; run < NRUNS; ++run) {
    km_init_kernel<<<NCP, 256, 0, stream>>>(tf, cent, run);
    for (int it = 0; it < NITERS; ++it) {
      km_prep_kernel<<<NCP, 64, 0, stream>>>(cent, centh, csq);
      km_assign_kernel<<<ND / 16, 256, 0, stream>>>(tf, centh, csq, assign);
      km_clear_kernel<<<(NC * DD + 255) / 256, 256, 0, stream>>>(sums, cnts);
      km_accum_kernel<<<(ND * 4) / 256, 256, 0, stream>>>(tea, inv_t, assign, sums, cnts);
      km_update_kernel<<<(NC * DD + 255) / 256, 256, 0, stream>>>(cent, sums, cnts);
    }
    km_prep_kernel<<<NCP, 64, 0, stream>>>(cent, centh, csq);
    km_assign_kernel<<<ND / 16, 256, 0, stream>>>(tf, centh, csq, labels + run * ND);
  }

  // 5) pos_mask (writes out region 1)
  finalize_kernel<<<(ND * TK) / 256, 256, 0, stream>>>(iknn, labels, ht, out);
}
